// FixedLearnableTensorSketch_21818433864279
// MI455X (gfx1250) — compile-verified, matrix operand order checked by reading, not run on hardware
//
#include <hip/hip_runtime.h>
#include <hip/hip_bf16.h>

typedef __attribute__((ext_vector_type(16))) _Float16 v16h;
typedef __attribute__((ext_vector_type(8)))  float    v8f;

union F16x8 { uint4 u; _Float16 h[8]; };

// Build a 16-element f16 WMMA fragment from two contiguous 8-halfword (16B) runs:
// elements [0..7] = base[kb .. kb+7], elements [8..15] = base[kb+16 .. kb+23]
__device__ __forceinline__ v16h make_frag(const _Float16* __restrict__ base, int kb) {
    F16x8 lo, hi;
    lo.u = *(const uint4*)(base + kb);
    hi.u = *(const uint4*)(base + kb + 16);
    v16h r;
#pragma unroll
    for (int i = 0; i < 8; ++i) { r[i] = lo.h[i]; r[i + 8] = hi.h[i]; }
    return r;
}

// One-shot conversion of W1/W2 (fp32, row-major [D,D]) to f16 in workspace.
__global__ __launch_bounds__(256) void cvt_w_kernel(const float* __restrict__ W1,
                                                    const float* __restrict__ W2,
                                                    _Float16* __restrict__ Wh, int n) {
    int i = blockIdx.x * blockDim.x + threadIdx.x;
    if (i < n)          Wh[i] = (_Float16)W1[i];
    else if (i < 2 * n) Wh[i] = (_Float16)W2[i - n];
}

__global__ __launch_bounds__(256) void sketch_fused_kernel(
    const int*      __restrict__ seq,     // [B, L], values in [0, 4)
    const float*    __restrict__ base,    // [B, D]
    const float*    __restrict__ cscale,  // [A]
    const float*    __restrict__ dimw,    // [D]
    const float*    __restrict__ bias,    // [D]
    const float*    __restrict__ chm,     // [A, D]
    const _Float16* __restrict__ W1h,     // [D, D] f16
    const float*    __restrict__ b1,      // [D]
    const _Float16* __restrict__ W2h,     // [D, D] f16
    const float*    __restrict__ b2,      // [D]
    float*          __restrict__ out,     // [B, D]
    int L)
{
    constexpr int D    = 128;
    constexpr int ROWS = 16;     // batch rows per workgroup
    constexpr int ESTR = 136;    // padded f16 row stride (16B-aligned: 272 bytes)

    __shared__ __align__(16) _Float16 eh[ROWS * ESTR];
    __shared__ __align__(16) _Float16 hh[ROWS * ESTR];
    __shared__ int   cpart[ROWS][16][4];
    __shared__ float counts[ROWS][4];

    const int tid = threadIdx.x;
    const int r0  = blockIdx.x * ROWS;

    // ---------------- Phase 1: per-row histogram (HBM-bound part) -------------
    // Values are 2-bit (A=4): accumulate moments instead of 4 compares/elem.
    //   sx = sum(x) = c1 + 2*c2 + 3*c3
    //   sb = sum(x>>1) = c2 + c3
    //   st = sum(x & (x>>1)) = c3        (only x==3 has both bits set)
    {
        const int row = tid >> 4;   // 16 threads per row
        const int sub = tid & 15;
        const int4* p = (const int4*)(seq + (size_t)(r0 + row) * L);
        const int nvec = L >> 2;
        int sx = 0, sb = 0, st = 0, nel = 0;
        for (int v = sub; v < nvec; v += 16) {   // coalesced 256B per row-group/iter
            __builtin_prefetch(p + v + 64, 0, 0);   // global_prefetch_b8, ~1KB ahead
            int4 x = p[v];
            int t;
            t = x.x >> 1; sx += x.x; sb += t; st += x.x & t;
            t = x.y >> 1; sx += x.y; sb += t; st += x.y & t;
            t = x.z >> 1; sx += x.z; sb += t; st += x.z & t;
            t = x.w >> 1; sx += x.w; sb += t; st += x.w & t;
            nel += 4;
        }
        const int c3 = st;
        const int c2 = sb - st;
        const int c1 = sx - 2 * c2 - 3 * c3;
        const int c0 = nel - c1 - c2 - c3;
        cpart[row][sub][0] = c0; cpart[row][sub][1] = c1;
        cpart[row][sub][2] = c2; cpart[row][sub][3] = c3;
    }
    __syncthreads();

    if (tid < 64) {   // reduce 16 partial counters per (row, char)
        int row = tid >> 2, a = tid & 3;
        int s = 0;
#pragma unroll
        for (int sub = 0; sub < 16; ++sub) s += cpart[row][sub][a];
        counts[row][a] = (float)s;
    }
    __syncthreads();

    // ---------------- Phase 2: build e tile (16 x 128) as f16 in LDS ----------
    {
        const int d   = tid & 127;
        const float dw = dimw[d], bs = bias[d];
        const float m0 = chm[0 * D + d], m1 = chm[1 * D + d];
        const float m2 = chm[2 * D + d], m3 = chm[3 * D + d];
        const float s0 = cscale[0], s1 = cscale[1], s2 = cscale[2], s3 = cscale[3];
        const float invL = 1.0f / (float)L;
#pragma unroll
        for (int rr = 0; rr < 8; ++rr) {
            int row = rr * 2 + (tid >> 7);
            float c0 = counts[row][0], c1 = counts[row][1];
            float c2 = counts[row][2], c3 = counts[row][3];
            float scl  = (c0 * s0 + c1 * s1 + c2 * s2 + c3 * s3) * invL;
            float mods = (c0 * m0 + c1 * m1 + c2 * m2 + c3 * m3) * invL;
            float ev = (base[(size_t)(r0 + row) * D + d] * dw + bs) * scl + mods;
            eh[row * ESTR + d] = (_Float16)ev;
        }
    }
    __syncthreads();

    // ---------------- Phase 3: h = relu(e @ W1^T + b1) via WMMA ---------------
    // 8 waves; wave w owns output columns [16w, 16w+16). K=128 in 4 steps of 32.
    const int wave  = tid >> 5;
    const int ln    = tid & 31;
    const int mrow  = ln & 15;                // A-fragment row
    const int khalf = (ln < 16) ? 0 : 8;      // K sub-run select / C row offset
    const int ncol  = wave * 16 + (ln & 15);  // B/C column

    v8f acc = {};
#pragma unroll
    for (int kt = 0; kt < 4; ++kt) {
        int kb = kt * 32 + khalf;
        v16h a = make_frag(eh + mrow * ESTR, kb);
        v16h b = make_frag(W1h + ncol * D, kb);   // B col n = W1 row n (e @ W1^T)
        acc = __builtin_amdgcn_wmma_f32_16x16x32_f16(
            false, a, false, b, (short)0, acc, false, false);
    }
    {
        const float bb = b1[ncol];
#pragma unroll
        for (int r = 0; r < 8; ++r) {
            int m = r + khalf;
            hh[m * ESTR + ncol] = (_Float16)fmaxf(acc[r] + bb, 0.0f);
        }
    }
    __syncthreads();

    // ---------------- Phase 4: out = h @ W2^T + b2 via WMMA -------------------
    v8f acc2 = {};
#pragma unroll
    for (int kt = 0; kt < 4; ++kt) {
        int kb = kt * 32 + khalf;
        v16h a = make_frag(hh + mrow * ESTR, kb);
        v16h b = make_frag(W2h + ncol * D, kb);
        acc2 = __builtin_amdgcn_wmma_f32_16x16x32_f16(
            false, a, false, b, (short)0, acc2, false, false);
    }
    {
        const float bb = b2[ncol];
#pragma unroll
        for (int r = 0; r < 8; ++r) {
            int m = r + khalf;
            out[(size_t)(r0 + m) * D + ncol] = acc2[r] + bb;
        }
    }
}

extern "C" void kernel_launch(void* const* d_in, const int* in_sizes, int n_in,
                              void* d_out, int out_size, void* d_ws, size_t ws_size,
                              hipStream_t stream) {
    const int*   seq    = (const int*)d_in[0];
    const float* base   = (const float*)d_in[1];
    const float* cscale = (const float*)d_in[2];
    const float* dimw   = (const float*)d_in[3];
    const float* bias   = (const float*)d_in[4];
    const float* chm    = (const float*)d_in[5];
    const float* W1     = (const float*)d_in[6];
    const float* b1     = (const float*)d_in[7];
    const float* W2     = (const float*)d_in[8];
    const float* b2     = (const float*)d_in[9];
    float*       out    = (float*)d_out;

    const int D  = in_sizes[3];           // 128
    const int Bn = in_sizes[1] / D;       // 4096
    const int L  = in_sizes[0] / Bn;      // 4096

    _Float16* W1h = (_Float16*)d_ws;
    _Float16* W2h = W1h + (size_t)D * D;

    const int n = D * D;
    cvt_w_kernel<<<(2 * n + 255) / 256, 256, 0, stream>>>(W1, W2, W1h, n);
    sketch_fused_kernel<<<Bn / 16, 256, 0, stream>>>(
        seq, base, cscale, dimw, bias, chm, W1h, b1, W2h, b2, out, L);
}
